// UBLSTMP_45079976739213
// MI455X (gfx1250) — matrix operand order repaired and checked by reading, not compiled, not run on hardware
//
#include <hip/hip_runtime.h>
#include <hip/hip_bf16.h>
#include <stdint.h>
#include <stddef.h>

// ---------------- problem constants (match reference) ----------------
#define B_    32
#define T_    512
#define D_    240
#define DP    256          // D padded to multiple of 32 (WMMA K)
#define H_    1024         // HF == HB
#define G4    4096         // 4*H
#define P_    512          // PF == PB
#define LBLK  32           // BWD_BLOCK
#define NB    (T_ / LBLK)  // 16 blocks
#define CLIPV 1.0f

typedef __bf16 bf16_t;
typedef __attribute__((ext_vector_type(16))) __bf16    v16bf;
typedef __attribute__((ext_vector_type(8)))  float     v8f;
typedef __attribute__((ext_vector_type(4)))  uint32_t  u32x4;

union FragU { v16bf v; u32x4 q[2]; };

// ---------------------------------------------------------------------
// bf16 WMMA fragment loaders (ISA 7.12.2 layouts, wave32).
// A (16x32, MxK): lanes 0-15 row M=lane hold K {0..7,16..23};
//                 lanes 16-31 row M=lane-16 hold K {8..15,24..31}.
// rowPtr must already point at element (row(lane), k0) for this lane.
template <typename T>
__device__ __forceinline__ v16bf loadFragA(const T* rowPtr, int lane) {
  const bool hi = (lane & 16) != 0;
  FragU f;
  f.q[0] = *(const u32x4*)(rowPtr + (hi ? 8 : 0));
  f.q[1] = *(const u32x4*)(rowPtr + (hi ? 24 : 16));
  return f.v;
}
// B (32x16, KxN): lane n<16 holds column n for K {0..7,16..23},
//                 lane n+16 holds column n for K {8..15,24..31}.
// colPtr points at element (k0, col(lane)); columns contiguous in K
// because our weights are stored [outCol][K] row-major.
template <typename T>
__device__ __forceinline__ v16bf loadFragB(const T* colPtr, int lane) {
  const bool hi = (lane & 16) != 0;
  FragU f;
  f.q[0] = *(const u32x4*)(colPtr + (hi ? 8 : 0));
  f.q[1] = *(const u32x4*)(colPtr + (hi ? 24 : 16));
  return f.v;
}

__device__ __forceinline__ float sigmoidf_(float x) {
  return 1.0f / (1.0f + __expf(-x));
}

#define WMMA_BF16(A, Bm, C) \
  __builtin_amdgcn_wmma_f32_16x16x32_bf16(false, (A), false, (Bm), (short)0, (C), false, false)

// ---------------------------------------------------------------------
// fp32 -> bf16 convert with column padding (zero fill).
__global__ void ublstmp_cvt_pad(const float* __restrict__ src,
                                bf16_t* __restrict__ dst,
                                int rows, int scols, int dcols) {
  int i = blockIdx.x * blockDim.x + threadIdx.x;
  int total = rows * dcols;
  if (i >= total) return;
  int r = i / dcols;
  int c = i - r * dcols;
  float v = (c < scols) ? src[(size_t)r * scols + c] : 0.0f;
  dst[i] = (bf16_t)v;
}

// ---------------------------------------------------------------------
// One LSTMP scan over `nsteps` for a 32-row batch tile.
// dir==0: rows are batch b=0..31, time t = s                (full fwd scan)
// dir==1: rows are (b*NB+blk) in [group*32, group*32+32),
//         time t = blk*LBLK + (LBLK-1) - s                  (block bwd scan)
// gates = [ r(512) | x_t(256) ] @ [ Wr | Wx ]^T  + bias     (fused input GEMM)
__device__ __forceinline__ void lstmp_scan(
    const bf16_t* __restrict__ dataBf,   // [B*T][DP] bf16 (zero padded)
    const float*  __restrict__ mask,     // [B][T]
    const bf16_t* __restrict__ wr,       // [4H][P]  bf16
    const bf16_t* __restrict__ wx,       // [4H][DP] bf16 (zero padded)
    const float*  __restrict__ bias,     // [4H]
    const bf16_t* __restrict__ wp,       // [P][H]   bf16
    float*        __restrict__ out,      // [B][T][2P]
    int nsteps, int dir, int group,
    bf16_t* __restrict__ rLds,           // [32][P]  bf16
    bf16_t* __restrict__ hLds)           // [32][H]  bf16
{
  const int tid  = threadIdx.x;
  const int lane = tid & 31;
  const int wave = tid >> 5;             // 16 waves
  const int lrow = lane & 15;
  const bool hi  = (lane & 16) != 0;

  // zero initial recurrent state
  for (int i = tid; i < B_ * P_; i += blockDim.x) rLds[i] = (bf16_t)0.0f;
  __syncthreads();

  // cell state privatized to registers: wave owns h-slice [wave*64, wave*64+64)
  float creg[2][4][8];
#pragma unroll
  for (int a = 0; a < 2; ++a)
#pragma unroll
    for (int b = 0; b < 4; ++b)
#pragma unroll
      for (int c = 0; c < 8; ++c) creg[a][b][c] = 0.0f;

  for (int s = 0; s < nsteps; ++s) {
    // -------- per-step row metadata (C-layout rows), hoisted out of tiles ---
    // VGPR j of an accumulator -> local row  mt*16 + (hi?8:0) + j
    float mval[2][8];
    int   outOff[2][8];
#pragma unroll
    for (int mt = 0; mt < 2; ++mt) {
#pragma unroll
      for (int j = 0; j < 8; ++j) {
        const int m    = mt * 16 + (hi ? 8 : 0) + j;
        const int grow = dir ? (group * 32 + m) : m;
        const int bb   = dir ? (grow >> 4) : m;        // NB == 16
        const int blk  = dir ? (grow & (NB - 1)) : 0;
        const int tt   = dir ? (blk * LBLK + (LBLK - 1) - s) : s;
        mval[mt][j]   = mask[bb * T_ + tt];
        outOff[mt][j] = (bb * T_ + tt) * (2 * P_) + (dir ? P_ : 0);
      }
    }

    // ================= gates + cell update =================
#pragma unroll
    for (int mt = 0; mt < 2; ++mt) {
      const int m0 = mt * 16;
      const int am = m0 + lrow;                       // A-fragment row (per lane)
      const int growA = dir ? (group * 32 + am) : am;
      const int bbA   = dir ? (growA >> 4) : am;
      const int blkA  = dir ? (growA & (NB - 1)) : 0;
      const int ttA   = dir ? (blkA * LBLK + (LBLK - 1) - s) : s;
      const bf16_t* xrow = dataBf + (size_t)(bbA * T_ + ttA) * DP;
      const bf16_t* rrow = rLds + (size_t)am * P_;

#pragma unroll
      for (int hc = 0; hc < 4; ++hc) {
        const int hbase = wave * 64 + hc * 16;
        const int bcol  = hbase + lrow;               // B-fragment column (per lane)
        v8f acc_i = {}, acc_f = {}, acc_g = {}, acc_o = {};

        // recurrent part: K = 512 over r (LDS, bf16).
        // unroll 2: enough ILP to overlap loads with WMMA, but keeps the
        // step-invariant weight fragments un-hoistable (no spills).
#pragma unroll 2
        for (int ks = 0; ks < 16; ++ks) {
          const int k0 = ks * 32;
          v16bf a = loadFragA(rrow + k0, lane);
          const bf16_t* wb = wr + (size_t)bcol * P_ + k0;
          __builtin_prefetch(wb + 32, 0, 1);          // global_prefetch next K chunk
          v16bf b0 = loadFragB(wb,                       lane);
          acc_i = WMMA_BF16(a, b0, acc_i);
          v16bf b1 = loadFragB(wb + (size_t)1 * H_ * P_, lane);
          acc_f = WMMA_BF16(a, b1, acc_f);
          v16bf b2 = loadFragB(wb + (size_t)2 * H_ * P_, lane);
          acc_g = WMMA_BF16(a, b2, acc_g);
          v16bf b3 = loadFragB(wb + (size_t)3 * H_ * P_, lane);
          acc_o = WMMA_BF16(a, b3, acc_o);
        }
        // fused input part: K = 256 over x_t (global, bf16, zero-padded)
#pragma unroll 2
        for (int ks = 0; ks < 8; ++ks) {
          const int k0 = ks * 32;
          v16bf a = loadFragA(xrow + k0, lane);
          const bf16_t* wb = wx + (size_t)bcol * DP + k0;
          v16bf b0 = loadFragB(wb,                       lane);
          acc_i = WMMA_BF16(a, b0, acc_i);
          v16bf b1 = loadFragB(wb + (size_t)1 * H_ * DP, lane);
          acc_f = WMMA_BF16(a, b1, acc_f);
          v16bf b2 = loadFragB(wb + (size_t)2 * H_ * DP, lane);
          acc_g = WMMA_BF16(a, b2, acc_g);
          v16bf b3 = loadFragB(wb + (size_t)3 * H_ * DP, lane);
          acc_o = WMMA_BF16(a, b3, acc_o);
        }

        // elementwise cell update on the C layout
        const int hcol = hbase + lrow;
        const float bi = bias[hcol];
        const float bf = bias[H_ + hcol];
        const float bg = bias[2 * H_ + hcol];
        const float bo = bias[3 * H_ + hcol];
#pragma unroll
        for (int j = 0; j < 8; ++j) {
          const int m  = m0 + (hi ? 8 : 0) + j;
          const float mv = mval[mt][j];
          const float gi = sigmoidf_(acc_i[j] + bi);
          const float gf = sigmoidf_(acc_f[j] + bf);
          const float gg = tanhf(acc_g[j] + bg);
          const float go = sigmoidf_(acc_o[j] + bo);
          const float cold = creg[mt][hc][j];
          const float cn = fminf(fmaxf(gf * cold + gi * gg, -CLIPV), CLIPV);
          creg[mt][hc][j] = mv * cn + (1.0f - mv) * cold;   // masked hold
          const float hv  = go * tanhf(cn);                 // pre-mask h (ref semantics)
          hLds[(size_t)m * H_ + hcol] = (bf16_t)hv;
        }
      }
    }
    __syncthreads();

    // ================= projection: r = h @ Wp^T =================
#pragma unroll
    for (int mt = 0; mt < 2; ++mt) {
      const int m0 = mt * 16;
      const bf16_t* hrow = hLds + (size_t)(m0 + lrow) * H_;
#pragma unroll
      for (int pc = 0; pc < 2; ++pc) {
        const int pbase = wave * 32 + pc * 16;
        v8f acc = {};
#pragma unroll 2
        for (int ks = 0; ks < 32; ++ks) {
          const int k0 = ks * 32;
          v16bf a = loadFragA(hrow + k0, lane);
          v16bf b = loadFragB(wp + (size_t)(pbase + lrow) * H_ + k0, lane);
          acc = WMMA_BF16(a, b, acc);
        }
        const int pcol = pbase + lrow;
#pragma unroll
        for (int j = 0; j < 8; ++j) {
          const int m  = m0 + (hi ? 8 : 0) + j;
          const float mv = mval[mt][j];
          const float rold = (float)rLds[(size_t)m * P_ + pcol];
          const float rv = mv * acc[j] + (1.0f - mv) * rold;
          out[(size_t)outOff[mt][j] + pcol] = rv;
          rLds[(size_t)m * P_ + pcol] = (bf16_t)rv;
        }
      }
    }
    __syncthreads();
  }
}

// ---------------------------------------------------------------------
// Grid: block 0 = full forward scan (T=512); blocks 1..16 = independent
// backward block-scans (32 rows of B*NB=512 each, 32 steps).
__global__ __launch_bounds__(512, 1) void ublstmp_scan_kernel(
    const bf16_t* __restrict__ dataBf, const float* __restrict__ mask,
    const bf16_t* __restrict__ wrF, const bf16_t* __restrict__ wxF,
    const float*  __restrict__ biasF, const bf16_t* __restrict__ wpF,
    const bf16_t* __restrict__ wrB, const bf16_t* __restrict__ wxB,
    const float*  __restrict__ biasB, const bf16_t* __restrict__ wpB,
    float* __restrict__ out) {
  extern __shared__ __align__(16) char smem[];
  bf16_t* rLds = (bf16_t*)smem;                                      // 32 KB
  bf16_t* hLds = (bf16_t*)(smem + (size_t)B_ * P_ * sizeof(bf16_t)); // 64 KB
  if (blockIdx.x == 0) {
    lstmp_scan(dataBf, mask, wrF, wxF, biasF, wpF, out, T_, 0, 0, rLds, hLds);
  } else {
    lstmp_scan(dataBf, mask, wrB, wxB, biasB, wpB, out, LBLK, 1,
               (int)blockIdx.x - 1, rLds, hLds);
  }
}

// ---------------------------------------------------------------------
extern "C" void kernel_launch(void* const* d_in, const int* in_sizes, int n_in,
                              void* d_out, int out_size, void* d_ws, size_t ws_size,
                              hipStream_t stream) {
  (void)in_sizes; (void)n_in; (void)out_size; (void)ws_size;

  const float* data   = (const float*)d_in[0];   // [B,T,D]
  const float* mask   = (const float*)d_in[1];   // [B,T]
  const float* wxF32  = (const float*)d_in[2];   // [4H,D]
  const float* wrF32  = (const float*)d_in[3];   // [4H,P]
  const float* biasF  = (const float*)d_in[4];   // [1,4H]
  const float* wpF32  = (const float*)d_in[5];   // [P,H]
  const float* wxB32  = (const float*)d_in[6];
  const float* wrB32  = (const float*)d_in[7];
  const float* biasB  = (const float*)d_in[8];
  const float* wpB32  = (const float*)d_in[9];
  float* out = (float*)d_out;

  // carve bf16 staging buffers out of workspace (256B aligned)
  size_t off = 0;
  auto carve = [&](size_t bytes) -> void* {
    void* p = (char*)d_ws + off;
    off += (bytes + 255) & ~(size_t)255;
    return p;
  };
  bf16_t* dataBf = (bf16_t*)carve((size_t)B_ * T_ * DP * 2);   // 8 MB
  bf16_t* wxF    = (bf16_t*)carve((size_t)G4 * DP * 2);        // 2 MB
  bf16_t* wrF    = (bf16_t*)carve((size_t)G4 * P_ * 2);        // 4 MB
  bf16_t* wpF    = (bf16_t*)carve((size_t)P_ * H_ * 2);        // 1 MB
  bf16_t* wxB    = (bf16_t*)carve((size_t)G4 * DP * 2);
  bf16_t* wrB    = (bf16_t*)carve((size_t)G4 * P_ * 2);
  bf16_t* wpB    = (bf16_t*)carve((size_t)P_ * H_ * 2);

  auto cvt = [&](const float* s, bf16_t* d, int rows, int sc, int dc) {
    int total = rows * dc;
    dim3 g((total + 255) / 256), b(256);
    hipLaunchKernelGGL(ublstmp_cvt_pad, g, b, 0, stream, s, d, rows, sc, dc);
  };
  cvt(data,  dataBf, B_ * T_, D_, DP);   // pad 240 -> 256 with zeros
  cvt(wxF32, wxF,    G4,      D_, DP);
  cvt(wrF32, wrF,    G4,      P_, P_);
  cvt(wpF32, wpF,    P_,      H_, H_);
  cvt(wxB32, wxB,    G4,      D_, DP);
  cvt(wrB32, wrB,    G4,      P_, P_);
  cvt(wpB32, wpB,    P_,      H_, H_);

  const size_t shmem = (size_t)B_ * P_ * 2 + (size_t)B_ * H_ * 2;  // 96 KB
  hipLaunchKernelGGL(ublstmp_scan_kernel, dim3(1 + NB), dim3(512), shmem, stream,
                     dataBf, mask, wrF, wxF, biasF, wpF,
                     wrB, wxB, biasB, wpB, out);
}
